// MultiHead_60748017434693
// MI455X (gfx1250) — matrix operand order repaired and checked by reading, not compiled, not run on hardware
//
#include <hip/hip_runtime.h>
#include <stdint.h>

typedef __bf16 v16bf __attribute__((ext_vector_type(16)));
typedef __bf16 v2bf  __attribute__((ext_vector_type(2)));
typedef float  v8f   __attribute__((ext_vector_type(8)));
typedef int    v4i   __attribute__((ext_vector_type(4)));

#define D_MODEL 1024
#define SEQ 2048
#define NB 4
#define NH 16
#define DH 64
#define INV_SCALE (1.0f/32.0f)   // 1/sqrt(1024)

#define AS1 __attribute__((address_space(1)))
#define AS3 __attribute__((address_space(3)))

#if __has_builtin(__builtin_amdgcn_global_load_async_to_lds_b128)
#define HAVE_ASYNC_LDS 1
#endif

static __device__ __forceinline__ void wait_async0(){
#if __has_builtin(__builtin_amdgcn_s_wait_asynccnt)
  __builtin_amdgcn_s_wait_asynccnt(0);
#else
  asm volatile("s_wait_asynccnt 0" ::: "memory");
#endif
}

// ---------- helpers ----------
static __device__ __forceinline__ unsigned short f2bf(float f){
  union { float f; uint32_t u; } v; v.f = f;
  uint32_t u = v.u;
  uint32_t r = (u + 0x7FFFu + ((u >> 16) & 1u)) >> 16;  // round-to-nearest-even
  return (unsigned short)r;
}
static __device__ __forceinline__ uint32_t pack2(float a, float b){
#if __has_builtin(__builtin_amdgcn_cvt_pk_bf16_f32)
  union { v2bf v; uint32_t u; } r;
  r.v = __builtin_amdgcn_cvt_pk_bf16_f32(a, b);
  return r.u;
#else
  return (uint32_t)f2bf(a) | ((uint32_t)f2bf(b) << 16);
#endif
}
static __device__ __forceinline__ v8f wmma_bf16(const uint32_t* a, const uint32_t* b, v8f c){
  union U { uint32_t u[8]; v16bf v; } ua, ub;
#pragma unroll
  for (int i = 0; i < 8; i++){ ua.u[i] = a[i]; ub.u[i] = b[i]; }
  return __builtin_amdgcn_wmma_f32_16x16x32_bf16(false, ua.v, false, ub.v,
                                                 (short)0, c, false, false);
}

// ---------- generic 128x128x32-tiled GEMM:  C = bf16(A) @ bf16(W) + bias ----------
// AF32: A is fp32 (convert while staging), else bf16.  OUTF32: store fp32, else bf16.
template<bool AF32, bool OUTF32>
static __device__ void gemm_tile(const void* __restrict__ Ap, const float* __restrict__ W,
                                 const float* __restrict__ bias, void* __restrict__ Cp,
                                 int K, int N)
{
  constexpr int LDA = 34;                       // even stride -> aligned u32 pair loads
  __shared__ unsigned short As[128 * 34];       // A tile, bf16, [m][k]
  __shared__ unsigned short Bt[128 * 34];       // W tile transposed, bf16, [n][k]

  const int tid  = threadIdx.x;
  const int lane = tid & 31, wave = tid >> 5;
  const int half = lane >> 4, l16 = lane & 15;
  const int wm = wave & 3, wn = wave >> 2;      // 4 waves on M x 2 waves on N
  const int m0 = blockIdx.y * 128, n0 = blockIdx.x * 128;

  v8f acc[2][4];
  v8f zf = 0.0f;
#pragma unroll
  for (int i = 0; i < 2; i++)
#pragma unroll
    for (int j = 0; j < 4; j++) acc[i][j] = zf;

  for (int kc = 0; kc < K; kc += 32) {
    // ---- stage A (128x32): 2 threads per row, 16 elems each ----
    {
      int row = tid >> 1;
      int col = (tid & 1) * 16;
      if (AF32) {
        const float* Ag = (const float*)Ap + (size_t)(m0 + row) * K + kc + col;
#pragma unroll
        for (int p = 0; p < 4; p++){
          float4 f = ((const float4*)Ag)[p];
          *(uint32_t*)&As[row * LDA + col + 4*p    ] = pack2(f.x, f.y);
          *(uint32_t*)&As[row * LDA + col + 4*p + 2] = pack2(f.z, f.w);
        }
      } else {
        const unsigned short* Ag = (const unsigned short*)Ap + (size_t)(m0 + row) * K + kc + col;
        uint4 u0 = ((const uint4*)Ag)[0];
        uint4 u1 = ((const uint4*)Ag)[1];
        uint32_t* dst = (uint32_t*)&As[row * LDA + col];
        dst[0]=u0.x; dst[1]=u0.y; dst[2]=u0.z; dst[3]=u0.w;
        dst[4]=u1.x; dst[5]=u1.y; dst[6]=u1.z; dst[7]=u1.w;
      }
    }
    // ---- stage W transposed (32x128 -> Bt[n][k]) ----
    {
      int krow = tid & 31;
      int nseg = (tid >> 5) * 16;
      const float* Wg = W + (size_t)(kc + krow) * N + n0 + nseg;
#pragma unroll
      for (int p = 0; p < 4; p++){
        float4 f = ((const float4*)Wg)[p];
        Bt[(nseg + 4*p + 0) * LDA + krow] = f2bf(f.x);
        Bt[(nseg + 4*p + 1) * LDA + krow] = f2bf(f.y);
        Bt[(nseg + 4*p + 2) * LDA + krow] = f2bf(f.z);
        Bt[(nseg + 4*p + 3) * LDA + krow] = f2bf(f.w);
      }
    }
    __syncthreads();

    // ---- fragments (ISA 7.12.2 16-bit layouts) ----
    uint32_t afr[2][8], bfr[4][8];
#pragma unroll
    for (int mt = 0; mt < 2; mt++){
      int row = wm*32 + mt*16 + l16;
#pragma unroll
      for (int j = 0; j < 8; j++){
        int k = (j >> 2)*16 + half*8 + 2*(j & 3);   // A: lanes0-15 K0-7/16-23, lanes16-31 K8-15/24-31
        afr[mt][j] = *(const uint32_t*)&As[row * LDA + k];
      }
    }
#pragma unroll
    for (int nt = 0; nt < 4; nt++){
      int n = wn*64 + nt*16 + l16;
#pragma unroll
      for (int j = 0; j < 8; j++){
        int k = (j >> 2)*16 + half*8 + 2*(j & 3);   // B: lanes0-15 K0-7/16-23, lanes16-31 K8-15/24-31
        bfr[nt][j] = *(const uint32_t*)&Bt[n * LDA + k];
      }
    }
#pragma unroll
    for (int mt = 0; mt < 2; mt++)
#pragma unroll
      for (int nt = 0; nt < 4; nt++)
        acc[mt][nt] = wmma_bf16(afr[mt], bfr[nt], acc[mt][nt]);
    __syncthreads();
  }

  // ---- epilogue: bias + store (C layout: slot r -> row r+8*half, lane -> col) ----
#pragma unroll
  for (int mt = 0; mt < 2; mt++){
#pragma unroll
    for (int nt = 0; nt < 4; nt++){
      int col = n0 + wn*64 + nt*16 + l16;
      float bvv = bias[col];
#pragma unroll
      for (int r = 0; r < 8; r++){
        int row = m0 + wm*32 + mt*16 + r + 8*half;
        float v = acc[mt][nt][r] + bvv;
        if (OUTF32) ((float*)Cp)[(size_t)row * N + col] = v;
        else        ((unsigned short*)Cp)[(size_t)row * N + col] = f2bf(v);
      }
    }
  }
}

__global__ __launch_bounds__(256) void proj_kernel(
    const float* x0, const float* x1, const float* x2,
    const float* w0, const float* w1, const float* w2,
    const float* b0, const float* b1, const float* b2,
    unsigned short* o0, unsigned short* o1, unsigned short* o2)
{
  const float* X; const float* W; const float* Bv; unsigned short* O;
  if (blockIdx.z == 0){ X = x0; W = w0; Bv = b0; O = o0; }
  else if (blockIdx.z == 1){ X = x1; W = w1; Bv = b1; O = o1; }
  else { X = x2; W = w2; Bv = b2; O = o2; }
  gemm_tile<true, false>(X, W, Bv, O, D_MODEL, D_MODEL);
}

__global__ __launch_bounds__(256) void outproj_kernel(
    const unsigned short* A, const float* W, const float* bias, float* Out)
{
  gemm_tile<false, true>(A, W, bias, Out, D_MODEL, D_MODEL);
}

// ---------- causal flash attention: one wave = one 16-query tile ----------
__global__ __launch_bounds__(256) void attn_kernel(const unsigned short* __restrict__ Qb,
                                                   const unsigned short* __restrict__ Kb,
                                                   const unsigned short* __restrict__ Vb,
                                                   unsigned short* __restrict__ Ob)
{
  constexpr int LKV = 72;                 // K/V chunk LDS row stride (mult of 8 -> b128)
  constexpr int LP  = 34;                 // P scratch stride (even -> u32 pair loads)
  __shared__ unsigned short Ks[32 * LKV];
  __shared__ unsigned short Vs[32 * LKV];
  __shared__ unsigned short Ps[8 * 16 * LP];

  const int tid  = threadIdx.x;
  const int lane = tid & 31, wave = tid >> 5;
  const int half = lane >> 4, l16 = lane & 15;
  const int b = blockIdx.x >> 4, h = blockIdx.x & 15;
  const int q0 = blockIdx.y * 128 + wave * 16;
  const size_t headoff = (size_t)h * DH;

  // Q fragment: A-layout, dh=64 split as two K=32 groups
  uint32_t qfr[2][8];
#pragma unroll
  for (int g = 0; g < 2; g++){
#pragma unroll
    for (int j = 0; j < 8; j++){
      int k = (j >> 2)*16 + half*8 + 2*(j & 3);
      size_t off = ((size_t)b*SEQ + q0 + l16) * D_MODEL + headoff + g*32 + k;
      qfr[g][j] = *(const uint32_t*)(Qb + off);
    }
  }

  v8f o[4]; v8f zf = 0.0f;
#pragma unroll
  for (int v = 0; v < 4; v++) o[v] = zf;
  float mrow[8], srow[8];
#pragma unroll
  for (int r = 0; r < 8; r++){ mrow[r] = -3.0e38f; srow[r] = 0.0f; }

  const int nchunks = blockIdx.y * 4 + 4;   // keys up to block's last query row
  for (int kc = 0; kc < nchunks; kc++){
    const int j0 = kc * 32;
    __syncthreads();
    {  // cooperative stage of 32-key K/V chunk (bf16, 8 elems = 16B per thread)
      int row = tid >> 3, seg = (tid & 7) * 8;
      size_t off = ((size_t)b*SEQ + j0 + row) * D_MODEL + headoff + seg;
#ifdef HAVE_ASYNC_LDS
      // ASYNCcnt-tracked memory->LDS DMA: no VGPR round-trip, waited on before use
      __builtin_amdgcn_global_load_async_to_lds_b128(
          (AS1 v4i*)(Kb + off), (AS3 v4i*)&Ks[row * LKV + seg], 0, 0);
      __builtin_amdgcn_global_load_async_to_lds_b128(
          (AS1 v4i*)(Vb + off), (AS3 v4i*)&Vs[row * LKV + seg], 0, 0);
      wait_async0();
#else
      *(uint4*)&Ks[row * LKV + seg] = *(const uint4*)(Kb + off);
      *(uint4*)&Vs[row * LKV + seg] = *(const uint4*)(Vb + off);
#endif
    }
    __syncthreads();
    if (j0 > q0 + 15) continue;            // fully-masked chunk for this wave

    // S = Q K^T  (two 16-key n-tiles, K accumulated over dh in 2 WMMAs each)
    v8f s[2]; s[0] = zf; s[1] = zf;
#pragma unroll
    for (int t = 0; t < 2; t++){
#pragma unroll
      for (int g = 0; g < 2; g++){
        uint32_t kfr[8];
#pragma unroll
        for (int j = 0; j < 8; j++){
          int k = (j >> 2)*16 + half*8 + 2*(j & 3);   // B = K^T: same k order as A
          kfr[j] = *(const uint32_t*)&Ks[(t*16 + l16) * LKV + g*32 + k];
        }
        s[t] = wmma_bf16(qfr[g], kfr, s[t]);
      }
    }
    // causal mask
#pragma unroll
    for (int t = 0; t < 2; t++){
      int j = j0 + t*16 + l16;
#pragma unroll
      for (int r = 0; r < 8; r++){
        int i = q0 + r + 8*half;
        if (j > i) s[t][r] = -1.0e30f;
      }
    }
    // online softmax (row = C-layout slot; reduce across 16-lane half)
#pragma unroll
    for (int r = 0; r < 8; r++){
      float x = fmaxf(s[0][r], s[1][r]);
#pragma unroll
      for (int m = 1; m < 16; m <<= 1) x = fmaxf(x, __shfl_xor(x, m, 32));
      float nm = fmaxf(mrow[r], x);
      float al = __expf((mrow[r] - nm) * INV_SCALE);
      mrow[r] = nm;
      float p0 = __expf((s[0][r] - nm) * INV_SCALE);
      float p1 = __expf((s[1][r] - nm) * INV_SCALE);
      s[0][r] = p0; s[1][r] = p1;
      float ps = p0 + p1;
#pragma unroll
      for (int m = 1; m < 16; m <<= 1) ps += __shfl_xor(ps, m, 32);
      srow[r] = srow[r] * al + ps;
#pragma unroll
      for (int v = 0; v < 4; v++) o[v][r] *= al;
    }
    // C-layout -> A-layout via per-wave LDS scratch (bf16)
    unsigned short* pw = &Ps[wave * 16 * LP];
#pragma unroll
    for (int t = 0; t < 2; t++)
#pragma unroll
      for (int r = 0; r < 8; r++)
        pw[(r + 8*half) * LP + t*16 + l16] = f2bf(s[t][r]);
    uint32_t pfr[8];
#pragma unroll
    for (int j = 0; j < 8; j++){
      int k = (j >> 2)*16 + half*8 + 2*(j & 3);
      pfr[j] = *(const uint32_t*)&pw[l16 * LP + k];
    }
    // O += P @ V  (4 n-tiles over dh, K=32 per WMMA; pack key-pairs manually)
#pragma unroll
    for (int v = 0; v < 4; v++){
      uint32_t vfr[8];
#pragma unroll
      for (int j = 0; j < 8; j++){
        int k = (j >> 2)*16 + half*8 + 2*(j & 3);
        uint32_t lo = Vs[(k    ) * LKV + v*16 + l16];
        uint32_t hi = Vs[(k + 1) * LKV + v*16 + l16];
        vfr[j] = lo | (hi << 16);
      }
      o[v] = wmma_bf16(pfr, vfr, o[v]);
    }
  }

  // finalize: divide by row sums, store bf16 [B,S,D] (heads concatenated)
#pragma unroll
  for (int r = 0; r < 8; r++){
    float inv = 1.0f / srow[r];
    int q = q0 + r + 8*half;
#pragma unroll
    for (int v = 0; v < 4; v++){
      size_t off = ((size_t)b*SEQ + q) * D_MODEL + headoff + v*16 + l16;
      Ob[off] = f2bf(o[v][r] * inv);
    }
  }
}

// ---------- host ----------
extern "C" void kernel_launch(void* const* d_in, const int* in_sizes, int n_in,
                              void* d_out, int out_size, void* d_ws, size_t ws_size,
                              hipStream_t stream) {
  const float* query = (const float*)d_in[0];
  const float* key   = (const float*)d_in[1];
  const float* value = (const float*)d_in[2];
  const float* wq = (const float*)d_in[3];
  const float* bq = (const float*)d_in[4];
  const float* wk = (const float*)d_in[5];
  const float* bk = (const float*)d_in[6];
  const float* wv = (const float*)d_in[7];
  const float* bv = (const float*)d_in[8];
  const float* wo = (const float*)d_in[9];
  const float* bo = (const float*)d_in[10];

  const size_t tokens = (size_t)NB * SEQ;       // 8192
  const size_t mat = tokens * D_MODEL;          // elements per [B*S, D] tensor
  unsigned short* Qb = (unsigned short*)d_ws;   // bf16 workspaces
  unsigned short* Kb = Qb + mat;
  unsigned short* Vb = Kb + mat;
  unsigned short* Ab = Vb + mat;

  dim3 blk(256, 1, 1);
  proj_kernel<<<dim3(D_MODEL/128, (unsigned)(tokens/128), 3), blk, 0, stream>>>(
      query, key, value, wq, wk, wv, bq, bk, bv, Qb, Kb, Vb);
  attn_kernel<<<dim3(NB*NH, SEQ/128, 1), blk, 0, stream>>>(Qb, Kb, Vb, Ab);
  outproj_kernel<<<dim3(D_MODEL/128, (unsigned)(tokens/128), 1), blk, 0, stream>>>(
      Ab, wo, bo, (float*)d_out);
}
